// SeparableAttn_8160437862502
// MI455X (gfx1250) — matrix operand order repaired and checked by reading, not compiled, not run on hardware
//
#include <hip/hip_runtime.h>
#include <stdint.h>

// ---------------- problem constants ----------------
#define NHEADS   12
#define KDIM     64
#define SEQ      1024
#define BATCH    64
#define CDIM     768
#define QKVO     1548      // (1 + 2*KD) * NH
#define HSTRIDE  129       // per-head stride inside qkv row

// ---------------- WMMA / TDM types ----------------
typedef __attribute__((ext_vector_type(16))) __bf16 v16bf;
typedef __attribute__((ext_vector_type(8)))  float  v8f;
typedef unsigned int u32x4 __attribute__((ext_vector_type(4)));
typedef int          i32x8 __attribute__((ext_vector_type(8)));
typedef int          i32x4 __attribute__((ext_vector_type(4)));

union FragAB { v16bf v; uint4 q[2]; };

#if defined(__has_builtin)
#  if __has_builtin(__builtin_amdgcn_tensor_load_to_lds) && \
      __has_builtin(__builtin_amdgcn_s_wait_tensorcnt)
#    define HAVE_TDM 1
#  endif
#endif
#ifndef HAVE_TDM
#  define HAVE_TDM 0
#endif

// f32 -> bf16 round-to-nearest-even
__device__ __forceinline__ unsigned short f2bf(float f) {
  union { float f; unsigned u; } c; c.f = f;
  unsigned u = c.u;
  u += 0x7FFFu + ((u >> 16) & 1u);
  return (unsigned short)(u >> 16);
}

// ---------------- 1) f32 -> bf16 convert ----------------
__global__ __launch_bounds__(256)
void cvt_bf16_kernel(const float* __restrict__ src,
                     unsigned short* __restrict__ dst, int n4) {
  int i = blockIdx.x * blockDim.x + threadIdx.x;
  if (i < n4) {
    float4 v = ((const float4*)src)[i];
    ushort4 o;
    o.x = f2bf(v.x); o.y = f2bf(v.y); o.z = f2bf(v.z); o.w = f2bf(v.w);
    ((ushort4*)dst)[i] = o;
  }
}

// ---------------- GEMM tiling ----------------
#define BM 128
#define BN 128
#define BK 64
#define LDST 72                       // padded LDS row stride (elems) = 144 B
#define TILE_ELEMS (BM * LDST)        // 9216 elems = 18432 B per tile buffer
#define GEMM_LDS_BYTES (4 * TILE_ELEMS * 2)  // A0,B0,A1,B1 = 73728 B

#if HAVE_TDM
// Low 32 bits of a flat LDS address are the LDS byte offset (aperture rule).
__device__ __forceinline__ unsigned lds_off(const void* p) {
  return (unsigned)(uintptr_t)p;
}

// Issue one TDM 2D tile load: tile = 128 rows x 64 bf16, row stride K elems,
// LDS rows padded by 16 B (pad_interval = 32 DWORDs, pad_amount = 4 DWORDs)
// so the landed layout matches LDST = 72 elems.
__device__ __forceinline__ void tdm_load_tile(unsigned lds_byte_off,
                                              const unsigned short* gptr,
                                              unsigned tdim0,   // row length (elems)
                                              unsigned tdim1,   // valid rows (OOB -> 0)
                                              unsigned stride0) // row stride (elems)
{
  const unsigned long long ga = (unsigned long long)(uintptr_t)gptr;
  u32x4 g0;
  g0[0] = 1u;                                              // count=1, user mode
  g0[1] = lds_byte_off;                                    // lds_addr
  g0[2] = (unsigned)ga;                                    // global_addr[31:0]
  g0[3] = (unsigned)((ga >> 32) & 0x01FFFFFFu) | (2u << 30); // addr[56:32] | type=2

  i32x8 g1;
  // workgroup_mask=0 | data_size=1 (2 B) | pad_enable | pad_interval=4 (32 DW)
  // | pad_amount=3 (4 DW)
  g1[0] = (int)((1u << 16) | (1u << 20) | (4u << 22) | (3u << 25));
  g1[1] = (int)((tdim0 & 0xFFFFu) << 16);                  // tensor_dim0[15:0]
  g1[2] = (int)(((tdim0 >> 16) & 0xFFFFu) | ((tdim1 & 0xFFFFu) << 16));
  g1[3] = (int)(((tdim1 >> 16) & 0xFFFFu) | ((unsigned)BK << 16)); // tile_dim0=64
  g1[4] = BM;                                              // tile_dim1=128, tile_dim2=0
  g1[5] = (int)stride0;                                    // tensor_dim0_stride[31:0]
  g1[6] = 0;                                               // stride0[47:32], stride1 lo
  g1[7] = 0;
  i32x4 z4 = {0, 0, 0, 0};
  i32x8 z8 = {0, 0, 0, 0, 0, 0, 0, 0};
  __builtin_amdgcn_tensor_load_to_lds(g0, g1, z4, z4, z8, 0);
}
#endif

// ---------------- 2/5) bf16 WMMA GEMM:  C = A[M,K] * W[N,K]^T + bias ----------------
__global__ __launch_bounds__(256)
void gemm_bf16_wmma_kernel(const unsigned short* __restrict__ A,   // M x K (bf16)
                           const unsigned short* __restrict__ W,   // N x K (bf16)
                           const float* __restrict__ bias,         // N
                           float* __restrict__ C,                  // M x N (f32)
                           int M, int N, int K) {
  extern __shared__ unsigned short smem[];   // [A0][B0][A1][B1]
  // NOTE: compute LDS buffer pointers at point of use (runtime GEP) — a
  // pointer array initialized with addrspacecast(LDS) breaks ld.lld.
  auto Atile = [&](int i) -> unsigned short* {
    return smem + (size_t)(i * 2) * TILE_ELEMS;
  };
  auto Btile = [&](int i) -> unsigned short* {
    return smem + (size_t)(i * 2 + 1) * TILE_ELEMS;
  };

  const int t     = threadIdx.x;
  const int wave  = t >> 5;
  const int lane  = t & 31;
  const int wm    = wave >> 1;          // 0..3 -> M sub-block of 32
  const int wn    = wave & 1;           // 0..1 -> N sub-block of 64
  const int lrow  = lane & 15;
  const int lhalf = lane >> 4;
  const int m0    = blockIdx.x * BM;
  const int n0    = blockIdx.y * BN;
  const int KT    = K / BK;

  v8f acc[2][4];
#pragma unroll
  for (int a = 0; a < 2; ++a)
#pragma unroll
    for (int b = 0; b < 4; ++b)
#pragma unroll
      for (int i = 0; i < 8; ++i) acc[a][b][i] = 0.f;

  // shared compute stage: 2 k-chunks of 32 from one LDS buffer pair
  auto compute_tile = [&](const unsigned short* As, const unsigned short* Bs) {
#pragma unroll
    for (int kc = 0; kc < BK; kc += 32) {
      FragAB fa[2], fb[4];
#pragma unroll
      for (int tm = 0; tm < 2; ++tm) {
        const unsigned short* base = As + (wm * 32 + tm * 16 + lrow) * LDST + kc;
        fa[tm].q[0] = *(const uint4*)(base + lhalf * 8);        // K = h*8..+7
        fa[tm].q[1] = *(const uint4*)(base + 16 + lhalf * 8);   // K = 16+h*8..+7
      }
#pragma unroll
      for (int tn = 0; tn < 4; ++tn) {
        const unsigned short* base =
            Bs + (wn * 64 + tn * 16 + lrow) * LDST + kc;
        fb[tn].q[0] = *(const uint4*)(base + lhalf * 8);        // K = h*8..+7
        fb[tn].q[1] = *(const uint4*)(base + 16 + lhalf * 8);   // K = 16+h*8..+7
      }
#pragma unroll
      for (int tm = 0; tm < 2; ++tm)
#pragma unroll
        for (int tn = 0; tn < 4; ++tn)
          acc[tm][tn] = __builtin_amdgcn_wmma_f32_16x16x32_bf16(
              false, fa[tm].v, false, fb[tn].v, (short)0, acc[tm][tn],
              false, false);
    }
  };

#if HAVE_TDM
  // ---- TDM double-buffered pipeline: wave 0 drives the Tensor Data Mover ----
  const bool issuer = (wave == 0);
  const size_t aBase = (size_t)m0 * K;
  const size_t bBase = (size_t)n0 * K;
  const unsigned arows = (unsigned)(M - m0);
  const unsigned brows = (N > n0) ? (unsigned)(N - n0) : 0u;

  if (issuer) {
    tdm_load_tile(lds_off(Atile(0)), A + aBase, (unsigned)K, arows, (unsigned)K);
    tdm_load_tile(lds_off(Btile(0)), W + bBase, (unsigned)K, brows, (unsigned)K);
  }
  for (int kt = 0; kt < KT; ++kt) {
    const int cur = kt & 1, nxt = cur ^ 1;
    if (issuer && (kt + 1 < KT)) {
      const size_t k1 = (size_t)(kt + 1) * BK;
      tdm_load_tile(lds_off(Atile(nxt)), A + aBase + k1, (unsigned)K, arows,
                    (unsigned)K);
      tdm_load_tile(lds_off(Btile(nxt)), W + bBase + k1, (unsigned)K, brows,
                    (unsigned)K);
    }
    if (issuer) {
      if (kt + 1 < KT) __builtin_amdgcn_s_wait_tensorcnt(2);  // tile kt landed
      else             __builtin_amdgcn_s_wait_tensorcnt(0);  // drain
    }
    __syncthreads();                 // release compute on buffer `cur`
    compute_tile(Atile(cur), Btile(cur));
    __syncthreads();                 // all reads done before `nxt` reuse
  }
#else
  // ---- fallback: VGPR-staged single buffer ----
  for (int k0 = 0; k0 < K; k0 += BK) {
#pragma unroll
    for (int s = t; s < BM * (BK / 8); s += 256) {
      const int r  = s >> 3;
      const int ck = (s & 7) * 8;
      uint4 va = *(const uint4*)(A + (size_t)(m0 + r) * K + k0 + ck);
      *(uint4*)(Atile(0) + r * LDST + ck) = va;
      uint4 vb = make_uint4(0u, 0u, 0u, 0u);
      if (n0 + r < N)
        vb = *(const uint4*)(W + (size_t)(n0 + r) * K + k0 + ck);
      *(uint4*)(Btile(0) + r * LDST + ck) = vb;
    }
    __syncthreads();
    compute_tile(Atile(0), Btile(0));
    __syncthreads();
  }
#endif

  // ---- store: C/D layout -> lane has (M = 8*lhalf + i, N = lrow) ----
#pragma unroll
  for (int tm = 0; tm < 2; ++tm) {
    const int row0 = m0 + wm * 32 + tm * 16 + lhalf * 8;
#pragma unroll
    for (int tn = 0; tn < 4; ++tn) {
      const int col = n0 + wn * 64 + tn * 16 + lrow;
      if (col < N) {
        const float bv = bias ? bias[col] : 0.f;
#pragma unroll
        for (int i = 0; i < 8; ++i)
          C[(size_t)(row0 + i) * N + col] = acc[tm][tn][i] + bv;
      }
    }
  }
}

// ---------------- 3) per-(b,h) softmax over N + cv reduction ----------------
__global__ __launch_bounds__(256)
void softmax_cv_kernel(const float* __restrict__ qkv,
                       const float* __restrict__ w_cv,
                       float* __restrict__ cv) {
  const int b = blockIdx.x / NHEADS;
  const int h = blockIdx.x % NHEADS;
  const int t = threadIdx.x;
  __shared__ float red[256];
  __shared__ float p[SEQ];

  const size_t baseq = (size_t)b * SEQ * QKVO + (size_t)h * HSTRIDE;

  float qv[4];
  float mx = -3.0e38f;
#pragma unroll
  for (int i = 0; i < 4; ++i) {
    const int n = t + i * 256;
    qv[i] = qkv[baseq + (size_t)n * QKVO];
    mx = fmaxf(mx, qv[i]);
  }
  red[t] = mx; __syncthreads();
  for (int s = 128; s > 0; s >>= 1) {
    if (t < s) red[t] = fmaxf(red[t], red[t + s]);
    __syncthreads();
  }
  mx = red[0]; __syncthreads();

  float sum = 0.f;
#pragma unroll
  for (int i = 0; i < 4; ++i) { qv[i] = __expf(qv[i] - mx); sum += qv[i]; }
  red[t] = sum; __syncthreads();
  for (int s = 128; s > 0; s >>= 1) {
    if (t < s) red[t] += red[t + s];
    __syncthreads();
  }
  const float inv = 1.f / red[0];
  __syncthreads();
#pragma unroll
  for (int i = 0; i < 4; ++i) {
    const int n = t + i * 256;
    p[n] = qv[i] * inv * w_cv[n];
  }
  __syncthreads();

  const int d = t & 63;
  const int g = t >> 6;
  const size_t basek = baseq + 1;
  float accd = 0.f;
  for (int n = g * 256; n < (g + 1) * 256; ++n)
    accd += p[n] * qkv[basek + (size_t)n * QKVO + d];
  red[t] = accd; __syncthreads();
  if (t < 64) {
    const float r = red[t] + red[t + 64] + red[t + 128] + red[t + 192];
    cv[((size_t)b * NHEADS + h) * KDIM + t] = r;
  }
}

// ---------------- 4) out_bf16 = relu(v) * cv ----------------
__global__ __launch_bounds__(256)
void relu_scale_kernel(const float* __restrict__ qkv,
                       const float* __restrict__ cv,
                       unsigned short* __restrict__ outb) {
  const size_t idx = (size_t)blockIdx.x * blockDim.x + threadIdx.x; // B*N*NH*16
  const int d4 = (int)(idx & 15);
  size_t rem = idx >> 4;
  const int h = (int)(rem % NHEADS); rem /= NHEADS;
  const int n = (int)(rem % SEQ);
  const int b = (int)(rem / SEQ);

  const size_t vbase =
      ((size_t)b * SEQ + n) * QKVO + (size_t)h * HSTRIDE + 1 + KDIM + d4 * 4;
  const size_t cbase = ((size_t)b * NHEADS + h) * KDIM + d4 * 4;

  ushort4 o;
  float v0 = qkv[vbase + 0], v1 = qkv[vbase + 1];
  float v2 = qkv[vbase + 2], v3 = qkv[vbase + 3];
  o.x = f2bf(fmaxf(v0, 0.f) * cv[cbase + 0]);
  o.y = f2bf(fmaxf(v1, 0.f) * cv[cbase + 1]);
  o.z = f2bf(fmaxf(v2, 0.f) * cv[cbase + 2]);
  o.w = f2bf(fmaxf(v3, 0.f) * cv[cbase + 3]);

  const size_t obase =
      (((size_t)b * SEQ + n) * (NHEADS * KDIM) + h * KDIM + d4 * 4);
  *(ushort4*)(outb + obase) = o;
}

// ---------------- launcher ----------------
extern "C" void kernel_launch(void* const* d_in, const int* in_sizes, int n_in,
                              void* d_out, int out_size, void* d_ws, size_t ws_size,
                              hipStream_t stream) {
  const float* x     = (const float*)d_in[0];   // 64*1024*768
  const float* w_qkv = (const float*)d_in[1];   // 1548*768
  const float* b_qkv = (const float*)d_in[2];   // 1548
  const float* w_cv  = (const float*)d_in[3];   // 1*1024
  const float* w_out = (const float*)d_in[4];   // 768*768
  const float* b_out = (const float*)d_in[5];   // 768
  float* y = (float*)d_out;                     // 64*1024*768

  const size_t M = (size_t)BATCH * SEQ;         // 65536

  // workspace layout (256-B aligned slices)
  char* ws = (char*)d_ws;
  size_t off = 0;
  auto carve = [&](size_t bytes) -> char* {
    char* p = ws + off;
    off = (off + bytes + 255) & ~(size_t)255;
    return p;
  };
  unsigned short* xb   = (unsigned short*)carve(M * CDIM * 2);
  unsigned short* wqb  = (unsigned short*)carve((size_t)QKVO * CDIM * 2);
  unsigned short* wob  = (unsigned short*)carve((size_t)CDIM * CDIM * 2);
  float*          qkv  = (float*)carve(M * QKVO * 4);
  float*          cv   = (float*)carve((size_t)BATCH * NHEADS * KDIM * 4);
  unsigned short* outb = (unsigned short*)carve(M * CDIM * 2);
  (void)ws_size;

  // 1) converts
  {
    int n4 = (int)(M * CDIM / 4);
    cvt_bf16_kernel<<<(n4 + 255) / 256, 256, 0, stream>>>(x, xb, n4);
    n4 = QKVO * CDIM / 4;
    cvt_bf16_kernel<<<(n4 + 255) / 256, 256, 0, stream>>>(w_qkv, wqb, n4);
    n4 = CDIM * CDIM / 4;
    cvt_bf16_kernel<<<(n4 + 255) / 256, 256, 0, stream>>>(w_out, wob, n4);
  }

  // 2) qkv = x @ w_qkv^T + b_qkv   (65536 x 1548, K=768)
  {
    dim3 grid((unsigned)(M / BM), (QKVO + BN - 1) / BN);
    gemm_bf16_wmma_kernel<<<grid, 256, GEMM_LDS_BYTES, stream>>>(
        xb, wqb, b_qkv, qkv, (int)M, QKVO, CDIM);
  }

  // 3) softmax over N + cv reduction
  softmax_cv_kernel<<<BATCH * NHEADS, 256, 0, stream>>>(qkv, w_cv, cv);

  // 4) out = relu(v) * cv  -> bf16
  {
    const size_t nthreads = M * NHEADS * 16;
    relu_scale_kernel<<<(unsigned)(nthreads / 256), 256, 0, stream>>>(qkv, cv,
                                                                      outb);
  }

  // 5) y = out @ w_out^T + b_out   (65536 x 768, K=768)
  {
    dim3 grid((unsigned)(M / BM), (CDIM + BN - 1) / BN);
    gemm_bf16_wmma_kernel<<<grid, 256, GEMM_LDS_BYTES, stream>>>(
        outb, wob, b_out, y, (int)M, CDIM, CDIM);
  }
}
